// CoFixing_73761768341673
// MI455X (gfx1250) — matrix operand order repaired and checked by reading, not compiled, not run on hardware
//
#include <hip/hip_runtime.h>
#include <hip/hip_fp16.h>
#include <math.h>

// ---------------------------------------------------------------------------
// CoFixing forward for MI455X (gfx1250, wave32, WMMA).
// The two 127x127 full-padding convs are exact dense GEMMs against Toeplitz
// matrices (each output pixel sums over the whole 64x64 input). We materialize
// the two weight matrices in f16 (2 x 64MB, L2-resident on the 192MB L2) in
// the layout Bt[n,k] so WMMA B fragments are contiguous b128 loads, and run
// both convs with v_wmma_f32_16x16x32_f16 (fp32 accumulate).
// Split-K=2 for occupancy (deterministic partial buffers). The GEMM K-loop is
// unrolled by 2 with two static fragment sets: loads of one set overlap WMMAs
// of the other (partial s_wait_loadcnt, no register-rotation moves).
// Workspace requirement: ~162 MB (guarded below, includes 4KB tail pad for
// the harmless dead prefetch-load of the last unrolled iteration).
// ---------------------------------------------------------------------------

typedef _Float16 v16h __attribute__((ext_vector_type(16)));
typedef _Float16 v8h  __attribute__((ext_vector_type(8)));
typedef float    v8f  __attribute__((ext_vector_type(8)));

#define NPIX 4096      // 64*64
#define NBV  16        // bs*nv
#define NJ   17
#define NL   16
#define NMAPJ (NBV*NJ) // 272 joint maps
#define NMAPL (NBV*NL) // 256 limb maps

// output section offsets (floats)
#define O_KPS   0          // (4,4,17,2,1)   = 544
#define O_COMBV 544        // (4,4,16,3)     = 768
#define O_COMBP 1312       // (4,4,16,2)     = 512
#define O_FNEW  1824       // (4,4,16,3,64,64)= 786432
#define O_HMS   788256     // (4,4,17,64,64) = 1114112

__constant__ int c_P0[16] = {0,1,2,0,4,5,0,7,8,9,8,11,12,8,14,15};
__constant__ int c_P1[16] = {1,2,3,4,5,6,7,8,9,10,11,12,13,14,15,16};
// for each joint j: limb index whose prox_ multiplies it (last l with P0[l]==j), or -1
__constant__ int c_jprox[17] = {6,1,2,-1,4,5,-1,7,13,9,-1,11,12,-1,14,15,-1};

// ---------------- block reductions (blockDim.x == 256, 8 waves) ------------
__device__ __forceinline__ float blockSum(float v, float* sm) {
  #pragma unroll
  for (int o = 16; o > 0; o >>= 1) v += __shfl_xor(v, o, 32);
  __syncthreads();
  if ((threadIdx.x & 31) == 0) sm[threadIdx.x >> 5] = v;
  __syncthreads();
  float t = 0.f;
  #pragma unroll
  for (int i = 0; i < 8; ++i) t += sm[i];
  return t;
}
__device__ __forceinline__ float blockMax(float v, float* sm) {
  #pragma unroll
  for (int o = 16; o > 0; o >>= 1) v = fmaxf(v, __shfl_xor(v, o, 32));
  __syncthreads();
  if ((threadIdx.x & 31) == 0) sm[threadIdx.x >> 5] = v;
  __syncthreads();
  float t = -3.0e38f;
  #pragma unroll
  for (int i = 0; i < 8; ++i) t = fmaxf(t, sm[i]);
  return t;
}

// ---------------- Toeplitz weight matrix generation ------------------------
// w1_c(dy,dx): u0 = 1-dx, u1 = 1-dy, den = u0^2+u1^2; w = (c? u1:u0)/den, 0 if den==0.
// mode 0: W1t[r = c*4096+p][s = q]        = w1_c(pos_p - pos_q)   (8192 x 4096)
// mode 1: W2t[r = p]       [s = c*4096+q] = w1_c(pos_q - pos_p)   (4096 x 8192)
__global__ void gen_w_kernel(_Float16* __restrict__ dst, int mode) {
  size_t base = ((size_t)blockIdx.x * blockDim.x + threadIdx.x) * 8;
  v8h o;
  if (mode == 0) {
    int r  = (int)(base >> 12);
    int s0 = (int)(base & 4095);
    int cch = r >> 12;
    int p = r & 4095, py = p >> 6, px = p & 63;
    #pragma unroll
    for (int e = 0; e < 8; ++e) {
      int q = s0 + e;
      int dy = py - (q >> 6), dx = px - (q & 63);
      int u0 = 1 - dx, u1 = 1 - dy;
      int den = u0 * u0 + u1 * u1;
      float w = den ? (float)(cch == 0 ? u0 : u1) / (float)den : 0.f;
      o[e] = (_Float16)w;
    }
  } else {
    int r  = (int)(base >> 13);
    int s0 = (int)(base & 8191);
    int py = r >> 6, px = r & 63;
    #pragma unroll
    for (int e = 0; e < 8; ++e) {
      int s = s0 + e;
      int cch = s >> 12;
      int q = s & 4095;
      int dy = (q >> 6) - py, dx = (q & 63) - px;
      int u0 = 1 - dx, u1 = 1 - dy;
      int den = u0 * u0 + u1 * u1;
      float w = den ? (float)(cch == 0 ? u0 : u1) / (float)den : 0.f;
      o[e] = (_Float16)w;
    }
  }
  *(v8h*)(dst + base) = o;
}

// ---------------- per-heatmap stats: center + std (ddof=1) -----------------
__global__ void hm_stats_kernel(const float* __restrict__ heat,
                                float* __restrict__ hmC,
                                float* __restrict__ stdhm) {
  __shared__ float sm[8];
  int map = blockIdx.x;
  const float* src = heat + (size_t)map * NPIX;
  float s = 0.f, s2 = 0.f;
  for (int i = threadIdx.x; i < NPIX; i += blockDim.x) { float x = src[i]; s += x; s2 += x * x; }
  float S = blockSum(s, sm), S2 = blockSum(s2, sm);
  float mean = S * (1.f / NPIX);
  float var  = fmaxf((S2 - S * mean) / (float)(NPIX - 1), 0.f);
  if (threadIdx.x == 0) stdhm[map] = sqrtf(var);
  float* dst = hmC + (size_t)map * NPIX;
  for (int i = threadIdx.x; i < NPIX; i += blockDim.x) dst[i] = src[i] - mean;
}

// ------- per-limb-map: std_df factor, comb_v (lvecs), comb_p (lpts) --------
__global__ void field_stats_kernel(const float* __restrict__ fields,
                                   float* __restrict__ factor,
                                   float* __restrict__ out,
                                   const int* __restrict__ ishape) {
  __shared__ float sm[8];
  int map = blockIdx.x;
  const float* F = fields + (size_t)map * 3 * NPIX;
  // std over all 3*4096 elements, ddof=1
  float s = 0.f, s2 = 0.f;
  for (int i = threadIdx.x; i < 3 * NPIX; i += blockDim.x) { float x = F[i]; s += x; s2 += x * x; }
  float S = blockSum(s, sm), S2 = blockSum(s2, sm);
  float mean = S / (float)(3 * NPIX);
  float sd = sqrtf(fmaxf((S2 - S * mean) / (float)(3 * NPIX - 1), 0.f));
  if (threadIdx.x == 0) factor[map] = sd / (sd + 1e-6f);
  // fnorm pass: sums for lvecs + max for softargmax
  float sv0 = 0.f, sv1 = 0.f, sv2 = 0.f, mx = -3.0e38f;
  for (int p = threadIdx.x; p < NPIX; p += blockDim.x) {
    float f0 = F[p], f1 = F[NPIX + p], f2 = F[2 * NPIX + p];
    float fn = sqrtf(f0 * f0 + f1 * f1 + f2 * f2);
    sv0 += fn * f0; sv1 += fn * f1; sv2 += fn * f2;
    mx = fmaxf(mx, fn);
  }
  sv0 = blockSum(sv0, sm); sv1 = blockSum(sv1, sm); sv2 = blockSum(sv2, sm);
  mx = blockMax(mx, sm);
  float se = 0.f, sr = 0.f, sc = 0.f;
  for (int p = threadIdx.x; p < NPIX; p += blockDim.x) {
    float f0 = F[p], f1 = F[NPIX + p], f2 = F[2 * NPIX + p];
    float fn = sqrtf(f0 * f0 + f1 * f1 + f2 * f2);
    float e = __expf(100.f * (fn - mx));
    se += e; sr += e * (float)(p >> 6); sc += e * (float)(p & 63);
  }
  se = blockSum(se, sm); sr = blockSum(sr, sm); sc = blockSum(sc, sm);
  if (threadIdx.x == 0) {
    float nrm = fmaxf(sqrtf(sv0 * sv0 + sv1 * sv1 + sv2 * sv2), 1e-12f);
    float* cv = out + O_COMBV + (size_t)map * 3;
    cv[0] = sv0 / nrm; cv[1] = sv1 / nrm; cv[2] = sv2 / nrm;
    float sx = (float)ishape[1] / 64.f, sy = (float)ishape[0] / 64.f;
    float* cp = out + O_COMBP + (size_t)map * 2;
    cp[0] = (sr / se) * sx; cp[1] = (sc / se) * sy;
  }
}

// ---------------- f16 operand packing --------------------------------------
__global__ void hs_build_kernel(const float* __restrict__ hmC, _Float16* __restrict__ Hs) {
  int idx = blockIdx.x * blockDim.x + threadIdx.x;  // 256*4096
  int map = idx >> 12, q = idx & 4095;
  int bv = map >> 4, l = map & 15;
  float a = hmC[(size_t)(bv * NJ + c_P1[l]) * NPIX + q];
  float b = hmC[(size_t)(bv * NJ + c_P0[l]) * NPIX + q];
  Hs[idx] = (_Float16)(a - b);
}

__global__ void fc_build_kernel(const float* __restrict__ fields,
                                const float* __restrict__ factor,
                                _Float16* __restrict__ fc) {
  int idx = blockIdx.x * blockDim.x + threadIdx.x;  // 256*8192
  int map = idx >> 13, rem = idx & 8191;
  int ch = rem >> 12, q = rem & 4095;
  fc[idx] = (_Float16)(fields[(size_t)map * 12288 + ch * 4096 + q] * factor[map]);
}

// ---------------- WMMA GEMM: C[M,N] = A[M,K] * Bt[N,K]^T, split-K ----------
// Each wave computes one 16(m) x 64(n) tile over K-range [sp*ksplit, ...).
// Split sp writes to its own partial C buffer. A,Bt row-major f16; C f32.
// K-loop unrolled by 2 with two static fragment register sets: loads of one
// set are issued before the WMMAs of the other, so waits are partial and the
// pipeline needs no register-rotation moves. The trailing dead load of the
// last iteration reads <=48 halves past kend (workspace has a tail pad).
#define LA(k)    __builtin_shufflevector(*(const v8h*)(Ar + (k)), *(const v8h*)(Ar + (k) + 16), \
                                         0,1,2,3,4,5,6,7,8,9,10,11,12,13,14,15)
#define LB(P, k) __builtin_shufflevector(*(const v8h*)((P) + (k)), *(const v8h*)((P) + (k) + 16), \
                                         0,1,2,3,4,5,6,7,8,9,10,11,12,13,14,15)
#define WMMAF16(a, b, c) __builtin_amdgcn_wmma_f32_16x16x32_f16(false, (a), false, (b), (short)0, (c), false, false)

__global__ void wmma_gemm_kernel(const _Float16* __restrict__ A, int lda,
                                 const _Float16* __restrict__ Bt, int ldb,
                                 float* __restrict__ C, int ldc,
                                 int mtiles, int nstrips, int ksplit) {
  int gtid = blockIdx.x * blockDim.x + threadIdx.x;
  int wave = gtid >> 5;
  int lane = gtid & 31;
  int hlf  = lane >> 4;   // 0: K 0-7,16-23 of each 32-K step, 1: K 8-15,24-31
  int lr   = lane & 15;
  int mt   = wave % mtiles;
  int rest = wave / mtiles;
  int ns   = rest % nstrips;
  int sp   = rest / nstrips;
  int k0   = sp * ksplit;
  int kend = k0 + ksplit;   // ksplit is a multiple of 64
  // A fragment: lane row = m (lr); halves e<8 -> K = hlf*8+e, e>=8 -> 16+hlf*8+(e-8)
  const _Float16* Ar = A  + (size_t)(mt * 16 + lr) * lda + hlf * 8;
  // B fragment: lane col = n (lr); halves e<8 -> K = hlf*8+e, e>=8 -> 16+hlf*8+(e-8)
  const _Float16* B0 = Bt + (size_t)(ns * 64 +  0 + lr) * ldb + hlf * 8;
  const _Float16* B1 = Bt + (size_t)(ns * 64 + 16 + lr) * ldb + hlf * 8;
  const _Float16* B2 = Bt + (size_t)(ns * 64 + 32 + lr) * ldb + hlf * 8;
  const _Float16* B3 = Bt + (size_t)(ns * 64 + 48 + lr) * ldb + hlf * 8;

  v8f acc0 = {}, acc1 = {}, acc2 = {}, acc3 = {};
  // prologue: set 0 at k0
  v16h a0  = LA(k0);
  v16h b00 = LB(B0, k0), b01 = LB(B1, k0), b02 = LB(B2, k0), b03 = LB(B3, k0);
  for (int k = k0; k < kend; k += 64) {
    int k1 = k + 32, k2 = k + 64;
    // issue set-1 loads; they overlap the set-0 WMMAs below
    v16h a1  = LA(k1);
    v16h b10 = LB(B0, k1), b11 = LB(B1, k1), b12 = LB(B2, k1), b13 = LB(B3, k1);
    acc0 = WMMAF16(a0, b00, acc0);
    acc1 = WMMAF16(a0, b01, acc1);
    acc2 = WMMAF16(a0, b02, acc2);
    acc3 = WMMAF16(a0, b03, acc3);
    // issue next-iteration set-0 loads; they overlap the set-1 WMMAs below
    // (dead on the final iteration; reads stay inside the padded workspace)
    a0  = LA(k2);
    b00 = LB(B0, k2); b01 = LB(B1, k2); b02 = LB(B2, k2); b03 = LB(B3, k2);
    acc0 = WMMAF16(a1, b10, acc0);
    acc1 = WMMAF16(a1, b11, acc1);
    acc2 = WMMAF16(a1, b12, acc2);
    acc3 = WMMAF16(a1, b13, acc3);
  }

  // C/D layout: lane L, vgpr r -> (M = r + 8*(L>=16), N = L%16)
  float* Cp = C + (size_t)sp * (size_t)(mtiles * 16) * ldc
                + (size_t)(mt * 16 + 8 * hlf) * ldc + ns * 64 + lr;
  #pragma unroll
  for (int r = 0; r < 8; ++r) {
    Cp[(size_t)r * ldc +  0] = acc0[r];
    Cp[(size_t)r * ldc + 16] = acc1[r];
    Cp[(size_t)r * ldc + 32] = acc2[r];
    Cp[(size_t)r * ldc + 48] = acc3[r];
  }
}

// ---------------- coef + fields_new (sums the 2 K-split partials) ----------
__global__ void coef_kernel(const float* __restrict__ out1,
                            const float* __restrict__ fields,
                            float* __restrict__ out) {
  int idx = blockIdx.x * blockDim.x + threadIdx.x;  // 256*4096
  int map = idx >> 12, p = idx & 4095;
  const size_t half2 = (size_t)NMAPL * 8192;  // partial-buffer stride
  float c0 = out1[(size_t)map * 8192 + p]        + out1[half2 + (size_t)map * 8192 + p];
  float c1 = out1[(size_t)map * 8192 + 4096 + p] + out1[half2 + (size_t)map * 8192 + 4096 + p];
  const float* F = fields + (size_t)map * 12288;
  float f0 = F[p], f1 = F[4096 + p], f2 = F[8192 + p];
  float coef = fmaxf(c0 * f0 + c1 * f1, 0.f);
  float* D = out + O_FNEW + (size_t)map * 12288;
  D[p] = f0 * coef; D[4096 + p] = f1 * coef; D[8192 + p] = f2 * coef;
}

// ---------------- final: hms_new, softargmax, keypoint select --------------
__global__ void final_kernel(const float* __restrict__ hmC, const float* __restrict__ stdhm,
                             const float* __restrict__ fds2jts, const float* __restrict__ confs,
                             const int* __restrict__ ishape,
                             const float* __restrict__ pth1p, const float* __restrict__ pth2p,
                             float* __restrict__ out) {
  __shared__ float sm[8];
  int map = blockIdx.x;
  int bv = map / NJ, j = map % NJ;
  const float* base = hmC + (size_t)map * NPIX;
  const size_t half2 = (size_t)NMAPL * NPIX;    // out2 partial-buffer stride
  int la = c_jprox[j];
  int lb = (j >= 1) ? (j - 1) : -1;
  const float* pa = (la >= 0) ? fds2jts + (size_t)(bv * NL + la) * NPIX : (const float*)0;
  const float* pb = (lb >= 0) ? fds2jts + (size_t)(bv * NL + lb) * NPIX : (const float*)0;
  float* hout = out + O_HMS + (size_t)map * NPIX;
  float s = 0.f, s2 = 0.f, mxm = -3.0e38f, mxc = -3.0e38f;
  for (int p = threadIdx.x; p < NPIX; p += blockDim.x) {
    float xc = base[p];
    mxc = fmaxf(mxc, xc);
    float x = xc;
    if (pa) x *= fmaxf(-(pa[p] + pa[half2 + p]), 0.f);   // prox_ = relu(-fds2jts)
    if (pb) x *= fmaxf( (pb[p] + pb[half2 + p]), 0.f);   // dist_ = relu(fds2jts)
    hout[p] = x;
    s += x; s2 += x * x; mxm = fmaxf(mxm, x);
  }
  float S = blockSum(s, sm), S2 = blockSum(s2, sm);
  float MXM = blockMax(mxm, sm), MXC = blockMax(mxc, sm);
  float mean = S * (1.f / NPIX);
  float var  = fmaxf((S2 - S * mean) / (float)(NPIX - 1), 0.f);
  float scale = stdhm[map] / (sqrtf(var) + 1e-6f);   // scale >= 0
  float mn = MXM * scale;
  float se1 = 0.f, sr1 = 0.f, sc1 = 0.f, se2 = 0.f, sr2 = 0.f, sc2 = 0.f;
  for (int p = threadIdx.x; p < NPIX; p += blockDim.x) {
    float e1 = __expf(100.f * (base[p] - MXC));
    float xn = hout[p] * scale;
    hout[p] = xn;                       // hms_new output (normalized)
    float e2 = __expf(100.f * (xn - mn));
    float row = (float)(p >> 6), col = (float)(p & 63);
    se1 += e1; sr1 += e1 * row; sc1 += e1 * col;
    se2 += e2; sr2 += e2 * row; sc2 += e2 * col;
  }
  se1 = blockSum(se1, sm); sr1 = blockSum(sr1, sm); sc1 = blockSum(sc1, sm);
  se2 = blockSum(se2, sm); sr2 = blockSum(sr2, sm); sc2 = blockSum(sc2, sm);
  if (threadIdx.x == 0) {
    float sx = (float)ishape[1] / 64.f, sy = (float)ishape[0] / 64.f;
    float k1x = (sr1 / se1) * sx, k1y = (sc1 / se1) * sy;
    float k2x = (sr2 / se2) * sx, k2y = (sc2 / se2) * sy;
    if (k2x != k2x) k2x = k1x;   // NaN fallback
    if (k2y != k2y) k2y = k1y;
    float dx = k2x - k1x, dy = k2y - k1y;
    float disp = sqrtf(dx * dx + dy * dy);
    float conf = confs[bv * (NL + NJ) + NL + j];
    int sel = (conf > pth1p[0]) && (disp < pth2p[0]);
    out[O_KPS + (size_t)map * 2 + 0] = sel ? k2x : k1x;
    out[O_KPS + (size_t)map * 2 + 1] = sel ? k2y : k1y;
  }
}

// ---------------------------------------------------------------------------
extern "C" void kernel_launch(void* const* d_in, const int* in_sizes, int n_in,
                              void* d_out, int out_size, void* d_ws, size_t ws_size,
                              hipStream_t stream) {
  const float* heat   = (const float*)d_in[0];
  const float* fields = (const float*)d_in[1];
  const int*   ishape = (const int*)d_in[5];
  const float* confs  = (const float*)d_in[6];
  const float* pth1   = (const float*)d_in[9];
  const float* pth2   = (const float*)d_in[10];
  float* out = (float*)d_out;

  // workspace layout (bytes): ~162 MB total (+4KB tail pad)
  char* ws = (char*)d_ws;
  _Float16* W1t = (_Float16*)ws;                              // 8192*4096 f16 = 64MB
  _Float16* W2t = W1t + (size_t)8192 * 4096;                  // 4096*8192 f16 = 64MB
  float* hmC    = (float*)(W2t + (size_t)4096 * 8192);        // 272*4096 f32
  float* stdhm  = hmC + (size_t)NMAPJ * NPIX;                 // 272 f32
  float* out1   = stdhm + NMAPJ;                              // 2 x 256*8192 f32 (split-K partials)
  float* out2   = out1 + 2 * (size_t)NMAPL * 8192;            // 2 x 256*4096 f32 (split-K partials)
  float* factor = out2 + 2 * (size_t)NMAPL * NPIX;            // 256 f32
  _Float16* Hs  = (_Float16*)(factor + NMAPL);                // 256*4096 f16
  _Float16* fc  = Hs + (size_t)NMAPL * NPIX;                  // 256*8192 f16
  size_t needed = (size_t)((char*)(fc + (size_t)NMAPL * 8192) - ws) + 4096 /*tail pad*/;
  if (ws_size < needed) return;  // insufficient scratch: bail deterministically

  // 1) Toeplitz weight matrices (L2-resident afterwards)
  gen_w_kernel<<<16384, 256, 0, stream>>>(W1t, 0);
  gen_w_kernel<<<16384, 256, 0, stream>>>(W2t, 1);
  // 2) per-map statistics
  hm_stats_kernel<<<NMAPJ, 256, 0, stream>>>(heat, hmC, stdhm);
  field_stats_kernel<<<NMAPL, 256, 0, stream>>>(fields, factor, out, ishape);
  // 3) f16 operand packing
  hs_build_kernel<<<(NMAPL * NPIX) / 256, 256, 0, stream>>>(hmC, Hs);
  fc_build_kernel<<<(NMAPL * 8192) / 256, 256, 0, stream>>>(fields, factor, fc);
  // 4) conv1 as WMMA GEMM: M=256, K=4096 (split 2x2048), N=8192
  //    waves = 16 mtiles * 128 nstrips * 2 splits = 4096 -> 512 blocks
  wmma_gemm_kernel<<<512, 256, 0, stream>>>(Hs, 4096, W1t, 4096, out1, 8192, 16, 128, 2048);
  // 5) coef + fields_new (adds the two partials)
  coef_kernel<<<(NMAPL * NPIX) / 256, 256, 0, stream>>>(out1, fields, out);
  // 6) conv2 as WMMA GEMM: M=256, K=8192 (split 2x4096), N=4096
  //    waves = 16 * 64 * 2 = 2048 -> 256 blocks
  wmma_gemm_kernel<<<256, 256, 0, stream>>>(fc, 8192, W2t, 8192, out2, 4096, 16, 64, 4096);
  // 7) hms_new + keypoints (adds the two partials)
  final_kernel<<<NMAPJ, 256, 0, stream>>>(hmC, stdhm, out2, confs, ishape, pth1, pth2, out);
}
